// GCN_82721070121690
// MI455X (gfx1250) — matrix operand order repaired and checked by reading, not compiled, not run on hardware
//
#include <hip/hip_runtime.h>

// ---------------- problem constants (match reference) ----------------
#define NNODES 50000
#define DEG    16
#define IN_F   128
#define HID    128
#define NCLS   64
#define BATCH  1024
#define SLEN   50

typedef __attribute__((ext_vector_type(16))) __bf16 v16bf;
typedef __attribute__((ext_vector_type(8)))  float  v8f;

struct U32B { uint4 lo, hi; };   // 32 bytes, bit-castable to v16bf

__device__ __forceinline__ unsigned short f2bf(float f) {
    unsigned int u = __float_as_uint(f);
    unsigned int r = u + 0x7FFFu + ((u >> 16) & 1u);   // round-to-nearest-even
    return (unsigned short)(r >> 16);
}
__device__ __forceinline__ float bf2f(unsigned short h) {
    return __uint_as_float(((unsigned int)h) << 16);
}

// ---------------------------------------------------------------
// Pack W [K x Nout] (fp32 row-major) into WMMA B-operand layout:
// tile (kt,nt): lane l holds 16 bf16: K = 32*kt + 16*(l/16) + 0..15,
// column n = 16*nt + (l%16).  Contiguous 32B per lane.
// ---------------------------------------------------------------
__global__ void pack_wmma_B(const float* __restrict__ W,
                            unsigned short* __restrict__ out,
                            int K, int Nout) {
    int tid = blockIdx.x * blockDim.x + threadIdx.x;
    int total = (K * Nout) >> 1;                 // each thread writes 2 bf16
    if (tid >= total) return;
    int Ntiles = Nout >> 4;
    int v    = tid & 7;
    int lane = (tid >> 3) & 31;
    int nt   = (tid >> 8) % Ntiles;
    int kt   = (tid >> 8) / Ntiles;
    int k = kt * 32 + ((lane >> 4) << 4) + 2 * v;
    int n = nt * 16 + (lane & 15);
    out[tid * 2 + 0] = f2bf(W[k * Nout + n]);
    out[tid * 2 + 1] = f2bf(W[(k + 1) * Nout + n]);
}

// ---------------------------------------------------------------
// mean aggregation over the fixed-degree edge list (fp32 in -> bf16 out)
// 32 lanes per node, 4 features per lane (float4 loads). 8 nodes/block.
// ---------------------------------------------------------------
__global__ void agg_mean_f32(const float* __restrict__ x,
                             const int* __restrict__ src,
                             unsigned short* __restrict__ out) {
    int node = blockIdx.x * 8 + (threadIdx.x >> 5);   // wave-uniform
    if (node >= NNODES) return;
    int lane = threadIdx.x & 31;
    int f0 = lane * 4;
    const int* sp = src + node * DEG;
    float4 acc = make_float4(0.f, 0.f, 0.f, 0.f);
#pragma unroll
    for (int k = 0; k < DEG; ++k) {
        int s = sp[k];
        float4 v = *reinterpret_cast<const float4*>(x + s * IN_F + f0);
        acc.x += v.x; acc.y += v.y; acc.z += v.z; acc.w += v.w;
    }
    const float sc = 1.0f / (float)DEG;
    ushort4 o;
    o.x = f2bf(acc.x * sc); o.y = f2bf(acc.y * sc);
    o.z = f2bf(acc.z * sc); o.w = f2bf(acc.w * sc);
    *reinterpret_cast<ushort4*>(out + node * IN_F + f0) = o;
}

// same, but bf16 input (layer-2 aggregation; halves gather traffic)
__global__ void agg_mean_bf16(const unsigned short* __restrict__ x,
                              const int* __restrict__ src,
                              unsigned short* __restrict__ out) {
    int node = blockIdx.x * 8 + (threadIdx.x >> 5);
    if (node >= NNODES) return;
    int lane = threadIdx.x & 31;
    int f0 = lane * 4;
    const int* sp = src + node * DEG;
    float a0 = 0.f, a1 = 0.f, a2 = 0.f, a3 = 0.f;
#pragma unroll
    for (int k = 0; k < DEG; ++k) {
        int s = sp[k];
        ushort4 v = *reinterpret_cast<const ushort4*>(x + s * HID + f0);
        a0 += bf2f(v.x); a1 += bf2f(v.y); a2 += bf2f(v.z); a3 += bf2f(v.w);
    }
    const float sc = 1.0f / (float)DEG;
    ushort4 o;
    o.x = f2bf(a0 * sc); o.y = f2bf(a1 * sc);
    o.z = f2bf(a2 * sc); o.w = f2bf(a3 * sc);
    *reinterpret_cast<ushort4*>(out + node * HID + f0) = o;
}

// ---------------------------------------------------------------
// WMMA GEMM: out = act(A[M, 32*KTILES](bf16 rowmajor) @ B + bias).
// Packed B is staged once per block into LDS (<=64KB, vs 320KB/WGP),
// then all 8 waves read it via ds_load_b128.  Each wave owns one
// 16-row M-tile and keeps up to 8 fp32 accumulators live so one
// A-fragment load feeds 8 independent v_wmma issues.
// ---------------------------------------------------------------
template <int KTILES, int NTILES, int RELU, int OUTF32, int ZERO0>
__global__ __launch_bounds__(256)
void wmma_gemm_bf16(const unsigned short* __restrict__ A,
                    const unsigned short* __restrict__ Bp,
                    const float* __restrict__ bias,
                    void* __restrict__ out, int M) {
    constexpr int KDIM = KTILES * 32;
    constexpr int NOUT = NTILES * 16;
    constexpr int BELEMS = KTILES * NTILES * 512;      // ushorts
    __shared__ __align__(16) unsigned short sB[BELEMS];

    // ---- cooperative stage of packed B into LDS (all 256 threads) ----
    {
        const uint4* g = reinterpret_cast<const uint4*>(Bp);
        uint4* s = reinterpret_cast<uint4*>(sB);
        constexpr int NV = BELEMS / 8;                 // uint4 count
#pragma unroll
        for (int i = 0; i < NV / 256; ++i)
            s[i * 256 + threadIdx.x] = g[i * 256 + threadIdx.x];
    }
    __syncthreads();

    int mt = blockIdx.x * 8 + (threadIdx.x >> 5);      // wave-uniform guard
    if (mt * 16 >= M) return;
    int lane = threadIdx.x & 31;
    int half = lane >> 4;
    const unsigned short* arow = A + (size_t)(mt * 16 + (lane & 15)) * KDIM;

    constexpr int CH = NTILES < 8 ? NTILES : 8;        // accumulators in flight
#pragma unroll
    for (int nc = 0; nc < NTILES; nc += CH) {
        v8f acc[CH];
#pragma unroll
        for (int i = 0; i < CH; ++i) acc[i] = {};
#pragma unroll
        for (int kt = 0; kt < KTILES; ++kt) {
            // A fragment: two b128 straight from row-major bf16
            U32B ua;
            ua.lo = *reinterpret_cast<const uint4*>(arow + kt * 32 + 16 * half);
            ua.hi = *reinterpret_cast<const uint4*>(arow + kt * 32 + 16 * half + 8);
            v16bf a = __builtin_bit_cast(v16bf, ua);
#pragma unroll
            for (int i = 0; i < CH; ++i) {
                const unsigned short* bb =
                    sB + ((size_t)((kt * NTILES) + (nc + i)) * 32 + lane) * 16;
                U32B ub;
                ub.lo = *reinterpret_cast<const uint4*>(bb);
                ub.hi = *reinterpret_cast<const uint4*>(bb + 8);
                v16bf b = __builtin_bit_cast(v16bf, ub);
                acc[i] = __builtin_amdgcn_wmma_f32_16x16x32_bf16(
                    false, a, false, b, (short)0, acc[i], false, false);
            }
        }
#pragma unroll
        for (int i = 0; i < CH; ++i) {
            int n = (nc + i) * 16 + (lane & 15);
            float bn = bias[n];
#pragma unroll
            for (int r = 0; r < 8; ++r) {
                int m = mt * 16 + r + 8 * half;        // C/D layout
                float v = acc[i][r] + bn;
                if (RELU)  v = fmaxf(v, 0.0f);
                if (ZERO0 && m == 0) v = 0.0f;
                if (OUTF32)
                    reinterpret_cast<float*>(out)[(size_t)m * NOUT + n] = v;
                else
                    reinterpret_cast<unsigned short*>(out)[(size_t)m * NOUT + n] = f2bf(v);
            }
        }
    }
}

// ---------------------------------------------------------------
// sentence embedding: sum 50 gathered rows of x2 [N,64] (f32) -> bf16
// ---------------------------------------------------------------
__global__ void sentence_sum(const float* __restrict__ x2,
                             const int* __restrict__ sentence,
                             unsigned short* __restrict__ outb) {
    int b = blockIdx.x;
    int f = threadIdx.x;          // 64 threads
    const int* sp = sentence + b * SLEN;
    float acc = 0.f;
#pragma unroll 5
    for (int l = 0; l < SLEN; ++l)
        acc += x2[(size_t)sp[l] * NCLS + f];
    outb[(size_t)b * NCLS + f] = f2bf(acc);
}

// final 128 -> 2 projection (too skinny for WMMA; plain dot)
__global__ void fc_final(const unsigned short* __restrict__ h2,
                         const float* __restrict__ Wf3,
                         const float* __restrict__ bf3,
                         float* __restrict__ out) {
    int t = blockIdx.x * blockDim.x + threadIdx.x;
    if (t >= BATCH * 2) return;
    int b = t >> 1, c = t & 1;
    const unsigned short* hr = h2 + (size_t)b * 128;
    float acc = 0.f;
#pragma unroll 8
    for (int k = 0; k < 128; ++k)
        acc += bf2f(hr[k]) * Wf3[k * 2 + c];
    out[t] = acc + bf3[c];
}

// ---------------------------------------------------------------
extern "C" void kernel_launch(void* const* d_in, const int* in_sizes, int n_in,
                              void* d_out, int out_size, void* d_ws, size_t ws_size,
                              hipStream_t stream) {
    const float* inputs = (const float*)d_in[0];
    const float* W1  = (const float*)d_in[1];
    const float* b1  = (const float*)d_in[2];
    const float* W2  = (const float*)d_in[3];
    const float* b2  = (const float*)d_in[4];
    const float* Wf1 = (const float*)d_in[5];
    const float* bf1 = (const float*)d_in[6];
    const float* Wf2 = (const float*)d_in[7];
    const float* bf2 = (const float*)d_in[8];
    const float* Wf3 = (const float*)d_in[9];
    const float* bf3 = (const float*)d_in[10];
    const int* src      = (const int*)d_in[11];
    // d_in[12] = dst: structurally repeat(arange(N),DEG); not needed.
    const int* sentence = (const int*)d_in[13];
    float* out = (float*)d_out;

    // ---- workspace layout (all sizes multiples of 256B) ----
    char* p = (char*)d_ws;
    unsigned short* aggA = (unsigned short*)p; p += (size_t)NNODES * IN_F * 2;  // 12.8MB
    unsigned short* x1   = (unsigned short*)p; p += (size_t)NNODES * HID * 2;   // 12.8MB
    unsigned short* aggX = (unsigned short*)p; p += (size_t)NNODES * HID * 2;   // 12.8MB
    float*          x2   = (float*)p;          p += (size_t)NNODES * NCLS * 4;  // 12.8MB
    unsigned short* sent = (unsigned short*)p; p += (size_t)BATCH * NCLS * 2;
    unsigned short* h1   = (unsigned short*)p; p += (size_t)BATCH * 256 * 2;
    unsigned short* h2   = (unsigned short*)p; p += (size_t)BATCH * 128 * 2;
    unsigned short* W1p  = (unsigned short*)p; p += (size_t)IN_F * HID * 2;
    unsigned short* W2p  = (unsigned short*)p; p += (size_t)HID * NCLS * 2;
    unsigned short* Wf1p = (unsigned short*)p; p += (size_t)NCLS * 256 * 2;
    unsigned short* Wf2p = (unsigned short*)p; p += (size_t)256 * 128 * 2;

    // ---- weight packing into WMMA B-operand layout ----
    pack_wmma_B<<<(IN_F * HID / 2 + 255) / 256, 256, 0, stream>>>(W1, W1p, IN_F, HID);
    pack_wmma_B<<<(HID * NCLS / 2 + 255) / 256, 256, 0, stream>>>(W2, W2p, HID, NCLS);
    pack_wmma_B<<<(NCLS * 256 / 2 + 255) / 256, 256, 0, stream>>>(Wf1, Wf1p, NCLS, 256);
    pack_wmma_B<<<(256 * 128 / 2 + 255) / 256, 256, 0, stream>>>(Wf2, Wf2p, 256, 128);

    // ---- layer 1: mean-agg + GEMM(relu) ----
    agg_mean_f32<<<NNODES / 8, 256, 0, stream>>>(inputs, src, aggA);
    wmma_gemm_bf16<4, 8, 1, 0, 0><<<(NNODES / 16 + 7) / 8, 256, 0, stream>>>(
        aggA, W1p, b1, x1, NNODES);

    // ---- layer 2: mean-agg + GEMM, x[0]=0 fused into store ----
    agg_mean_bf16<<<NNODES / 8, 256, 0, stream>>>(x1, src, aggX);
    wmma_gemm_bf16<4, 4, 0, 1, 1><<<(NNODES / 16 + 7) / 8, 256, 0, stream>>>(
        aggX, W2p, b2, x2, NNODES);

    // ---- sentence embedding + MLP head ----
    sentence_sum<<<BATCH, NCLS, 0, stream>>>(x2, sentence, sent);
    wmma_gemm_bf16<2, 16, 1, 0, 0><<<(BATCH / 16 + 7) / 8, 256, 0, stream>>>(
        sent, Wf1p, bf1, h1, BATCH);
    wmma_gemm_bf16<8, 8, 1, 0, 0><<<(BATCH / 16 + 7) / 8, 256, 0, stream>>>(
        h1, Wf2p, bf2, h2, BATCH);
    fc_final<<<(BATCH * 2 + 255) / 256, 256, 0, stream>>>(h2, Wf3, bf3, out);
}
